// WindowMHA2d_25331717112565
// MI455X (gfx1250) — matrix operand order repaired and checked by reading, not compiled, not run on hardware
//
#include <hip/hip_runtime.h>
#include <hip/hip_bf16.h>

typedef __attribute__((ext_vector_type(16))) _Float16 v16h;
typedef __attribute__((ext_vector_type(8)))  _Float16 v8h;
typedef __attribute__((ext_vector_type(8)))  float    v8f;
typedef __attribute__((ext_vector_type(4)))  float    v4f;

// Problem constants (match reference setup_inputs)
constexpr int CB    = 4;              // batch
constexpr int C     = 192;            // channels
constexpr int HH    = 384;
constexpr int WW    = 384;
constexpr int SH    = 8, SW = 8;      // window
constexpr int NT    = 64;             // tokens per window
constexpr int NHEAD = 6;
constexpr int DH    = 32;             // head dim
constexpr int OHW   = HH / SH;        // 48
constexpr int OWW   = WW / SW;        // 48
constexpr int NWIN  = CB * OHW * OWW; // 9216

constexpr int QKV_ROWS  = 3 * C;          // 576
constexpr int NQKV_W    = QKV_ROWS * C;   // 110592
constexpr int NHEAD_W   = C * C;          // 36864

// LDS pitches (halves / floats), multiples of 8 -> 16B-aligned fragment loads
constexpr int XO_P = 200;   // X (input, later O) : [64][200] f16
constexpr int QK_P = 200;   // Q, K               : [64][200] f16
constexpr int VT_P = 72;    // V transposed       : [192][72] f16
constexpr int S_P  = 66;    // scores             : [64][66] f32
constexpr int AT_P = 72;    // attn probs         : [64][72] f16

union H16 { v16h v; v8h h[2]; _Float16 e[16]; };

// A-fragment (16x32 f16, M rows striped per lane) from row-major f16 LDS.
// lane<16 : a[0..7]=K(kbase+0..7),   a[8..15]=K(kbase+8..15)
// lane>=16: a[0..7]=K(kbase+16..23), a[8..15]=K(kbase+24..31)
__device__ inline v16h load_a_frag(const _Float16* row, int kbase, int hf) {
  H16 a;
  a.h[0] = *(const v8h*)(row + kbase + 16 * hf);
  a.h[1] = *(const v8h*)(row + kbase + 8 + 16 * hf);
  return a.v;
}

// B-fragment (32x16 f16, N cols striped per lane) from any [col][K] f16 array
// (LDS or global): lane holds 16 contiguous K values starting at kbase + 16*hf.
__device__ inline v16h load_b_f16(const _Float16* colrow, int kbase, int hf) {
  H16 b;
  const v8h* p = (const v8h*)(colrow + kbase + 16 * hf);
  b.h[0] = p[0];
  b.h[1] = p[1];
  return b.v;
}

// ---- Pre-pass: convert f32 weights -> f16 once (L2-resident, reused by all
// 9216 window blocks; removes per-tile v_cvt chains from the hot loops). ----
__global__ __launch_bounds__(256) void cvt_weights(
    const float* __restrict__ qkv_w, const float* __restrict__ head_w,
    _Float16* __restrict__ wq, _Float16* __restrict__ wh)
{
  const int i = blockIdx.x * 256 + threadIdx.x;
  if (i < NQKV_W)  wq[i] = (_Float16)qkv_w[i];
  if (i < NHEAD_W) wh[i] = (_Float16)head_w[i];
}

__global__ __launch_bounds__(256) void win_mha_fused(
    const float* __restrict__ x,
    const _Float16* __restrict__ wq,  const float* __restrict__ qkv_b,
    const _Float16* __restrict__ wh,  const float* __restrict__ head_b,
    float* __restrict__ out)
{
  __shared__ _Float16 s_xo[NT * XO_P];  // X, later reused as O
  __shared__ _Float16 s_q [NT * QK_P];
  __shared__ _Float16 s_k [NT * QK_P];
  __shared__ _Float16 s_vt[C  * VT_P];  // V transposed: [d_global][token]
  __shared__ float    s_s [NT * S_P];
  __shared__ _Float16 s_at[NT * AT_P];

  const int tid  = threadIdx.x;
  const int lane = tid & 31;
  const int wv   = tid >> 5;     // wave id 0..7
  const int n16  = lane & 15;    // N / M index within tile
  const int hf   = lane >> 4;    // half-select

  const int wi = blockIdx.x;
  const int b  = wi / (OHW * OWW);
  const int rr = wi % (OHW * OWW);
  const int h0 = (rr / OWW) * SH;
  const int w0 = (rr % OWW) * SW;

  // ---- Phase 0: load window (f32 strided BCHW) -> LDS f16 token-major ----
  for (int s = tid; s < C * SH; s += 256) {       // 1536 row-segments of 8 px
    const int c  = s >> 3;
    const int ty = s & 7;
    const float* src = x + (((size_t)(b * C + c) * HH + h0 + ty) * WW + w0);
    const v4f* src4 = (const v4f*)src;
    v4f p0 = src4[0], p1 = src4[1];
#pragma unroll
    for (int tx = 0; tx < 4; ++tx) {
      s_xo[(ty * 8 + tx)     * XO_P + c] = (_Float16)p0[tx];
      s_xo[(ty * 8 + tx + 4) * XO_P + c] = (_Float16)p1[tx];
    }
  }
  __syncthreads();

  // ---- Phase 1: QKV = X @ qkv_w^T + b  (64x576, K=192) ----
  // Fixed M-tile per wave (2 waves per tm); A fragments for all 6 K-steps
  // stay in registers across the 18 N-tiles each wave sweeps.
  {
    const int tm = wv & 3;
    v16h afr[6];
#pragma unroll
    for (int kk = 0; kk < 6; ++kk)
      afr[kk] = load_a_frag(&s_xo[(tm * 16 + n16) * XO_P], kk * 32, hf);
    const int t0 = tm * 16 + hf * 8;

    for (int tn = (wv >> 2); tn < 36; tn += 2) {
      const int j = tn * 16 + n16;          // 0..575
      const _Float16* wrow = wq + (size_t)j * C;
      v8f acc = {};
#pragma unroll
      for (int kk = 0; kk < 6; ++kk) {
        v16h bm = load_b_f16(wrow, kk * 32, hf);
        acc = __builtin_amdgcn_wmma_f32_16x16x32_f16(false, afr[kk], false, bm,
                                                     (short)0, acc, false, false);
      }
      const float bias = qkv_b[j];
      // Region boundaries (192/384) are multiples of 16, so the whole wave is
      // in one region per tile; force a truly scalar branch on tn.
      const int tnu = __builtin_amdgcn_readfirstlane(tn);
      if (tnu < 12) {                    // Q row-major
#pragma unroll
        for (int r = 0; r < 8; ++r)
          s_q[(t0 + r) * QK_P + j] = (_Float16)(acc[r] + bias);
      } else if (tnu < 24) {             // K row-major
#pragma unroll
        for (int r = 0; r < 8; ++r)
          s_k[(t0 + r) * QK_P + (j - C)] = (_Float16)(acc[r] + bias);
      } else {                           // V transposed: [d][token]
        v8h pk;
#pragma unroll
        for (int r = 0; r < 8; ++r) pk[r] = (_Float16)(acc[r] + bias);
        *(v8h*)&s_vt[(j - 2 * C) * VT_P + t0] = pk;
      }
    }
  }
  __syncthreads();

  // ---- Phase 2: per-head attention ----
  const float scale = 0.17677669529663687f;  // 1/sqrt(32)
  for (int h = 0; h < NHEAD; ++h) {
    // scores = Q_h @ K_h^T * scale : 16 tiles, one 16x16x32 WMMA each.
    // Fixed tm per wave -> single A fragment reused for both N-tiles.
    {
      const int tm = wv >> 1;
      v16h a = load_a_frag(&s_q[(tm * 16 + n16) * QK_P], h * DH, hf);
      const int t0 = tm * 16 + hf * 8;
#pragma unroll
      for (int i = 0; i < 2; ++i) {
        const int tn = (wv & 1) * 2 + i;
        v16h bm = load_b_f16(&s_k[(tn * 16 + n16) * QK_P], h * DH, hf);
        v8f acc = {};
        acc = __builtin_amdgcn_wmma_f32_16x16x32_f16(false, a, false, bm,
                                                     (short)0, acc, false, false);
#pragma unroll
        for (int r = 0; r < 8; ++r)
          s_s[(t0 + r) * S_P + tn * 16 + n16] = acc[r] * scale;
      }
    }
    __syncthreads();

    // row softmax (64 rows; passes through LDS keep VGPR use low)
    if (tid < NT) {
      float* srow = &s_s[tid * S_P];
      float m = -1e30f;
      for (int n = 0; n < NT; ++n) m = fmaxf(m, srow[n]);
      float sum = 0.f;
      for (int n = 0; n < NT; ++n) {
        float e = __expf(srow[n] - m);
        srow[n] = e;
        sum += e;
      }
      const float inv = 1.0f / sum;
      for (int n = 0; n < NT; ++n)
        s_at[tid * AT_P + n] = (_Float16)(srow[n] * inv);
    }
    __syncthreads();

    // O_h = attn @ V_h : 4x2 tiles, 2 K-steps (K=64); one tile per wave
    {
      const int tm = wv & 3;
      const int tn = wv >> 2;
      v8f acc = {};
#pragma unroll
      for (int kk = 0; kk < 2; ++kk) {
        v16h a  = load_a_frag(&s_at[(tm * 16 + n16) * AT_P], kk * 32, hf);
        v16h bm = load_b_f16(&s_vt[(h * DH + tn * 16 + n16) * VT_P], kk * 32, hf);
        acc = __builtin_amdgcn_wmma_f32_16x16x32_f16(false, a, false, bm,
                                                     (short)0, acc, false, false);
      }
      const int t0 = tm * 16 + hf * 8;
#pragma unroll
      for (int r = 0; r < 8; ++r)
        s_xo[(t0 + r) * XO_P + h * DH + tn * 16 + n16] = (_Float16)acc[r];
    }
    __syncthreads();
  }

  // ---- Phase 3: Y = O @ head_w^T + b, scatter back to BCHW (f32) ----
  // Same A-register-resident structure as QKV.
  {
    const int tm = wv & 3;
    v16h afr[6];
#pragma unroll
    for (int kk = 0; kk < 6; ++kk)
      afr[kk] = load_a_frag(&s_xo[(tm * 16 + n16) * XO_P], kk * 32, hf);
    const int t0 = tm * 16 + hf * 8;

    for (int tn = (wv >> 2); tn < 12; tn += 2) {
      const int cj = tn * 16 + n16;      // output channel
      const _Float16* wrow = wh + (size_t)cj * C;
      v8f acc = {};
#pragma unroll
      for (int kk = 0; kk < 6; ++kk) {
        v16h bm = load_b_f16(wrow, kk * 32, hf);
        acc = __builtin_amdgcn_wmma_f32_16x16x32_f16(false, afr[kk], false, bm,
                                                     (short)0, acc, false, false);
      }
      const float bias = head_b[cj];
#pragma unroll
      for (int r = 0; r < 8; ++r) {
        const int t = t0 + r;
        out[((size_t)(b * C + cj) * HH + h0 + (t >> 3)) * WW + w0 + (t & 7)]
            = acc[r] + bias;
      }
    }
  }
}

extern "C" void kernel_launch(void* const* d_in, const int* in_sizes, int n_in,
                              void* d_out, int out_size, void* d_ws, size_t ws_size,
                              hipStream_t stream) {
  const float* x      = (const float*)d_in[0];
  const float* qkv_w  = (const float*)d_in[1];
  const float* qkv_b  = (const float*)d_in[2];
  const float* head_w = (const float*)d_in[3];
  const float* head_b = (const float*)d_in[4];
  float* out = (float*)d_out;

  // f16 weight staging in workspace (qkv: 110592 halves, head: 36864 halves)
  _Float16* wq = (_Float16*)d_ws;
  _Float16* wh = wq + NQKV_W;

  const int cvt_blocks = (NQKV_W + 255) / 256;   // covers head_w too
  cvt_weights<<<cvt_blocks, 256, 0, stream>>>(qkv_w, head_w, wq, wh);
  win_mha_fused<<<NWIN, 256, 0, stream>>>(x, wq, qkv_b, wh, head_b, out);
}
